// TimeSelfAttentionFallback_62895501082759
// MI455X (gfx1250) — matrix operand order repaired and checked by reading, not compiled, not run on hardware
//
#include <hip/hip_runtime.h>

#define NHEADS 12
#define DK     64
#define DMODEL 768
#define BATCH  2
#define SEQ    2048
#define MTOT   (BATCH * SEQ)   // 4096

typedef __attribute__((ext_vector_type(16))) __bf16 v16bf;
typedef __attribute__((ext_vector_type(8)))  __bf16 v8bf;
typedef __attribute__((ext_vector_type(8)))  float  v8f;

// ---------- bf16 helpers (round-to-nearest-even) ----------
__device__ __forceinline__ __bf16 f2bf(float x) {
    unsigned u = __builtin_bit_cast(unsigned, x);
    unsigned r = (u + 0x7FFFu + ((u >> 16) & 1u)) >> 16;
    unsigned short s = (unsigned short)r;
    return __builtin_bit_cast(__bf16, s);
}
__device__ __forceinline__ float bf2f(__bf16 x) {
    unsigned short s = __builtin_bit_cast(unsigned short, x);
    unsigned u = ((unsigned)s) << 16;
    return __builtin_bit_cast(float, u);
}

// ---------- CDNA5 async LDS DMA (ISA §15.18.3 op 98, ASYNCcnt-tracked) ----------
// LDS aperture: low 32 bits of a generic shared pointer are the LDS address.
__device__ __forceinline__ void async_b128(const void* gptr, void* ldsptr) {
    unsigned l = (unsigned)(unsigned long long)ldsptr;
    asm volatile("global_load_async_to_lds_b128 %0, %1, off"
                 :: "v"(l), "v"(gptr) : "memory");
}
__device__ __forceinline__ void wait_async0() {
    asm volatile("s_wait_asynccnt 0" ::: "memory");
}
__device__ __forceinline__ void wait_async4() {
    asm volatile("s_wait_asynccnt 4" ::: "memory");   // retire previous tile's 4
}

// ---------- WMMA fragment helpers (contiguous K order used by this code, wave32) ----------
// A-matrix 16x32 bf16: lane m=lane&15 holds row m. lane<16: elems 0..15 = K0..15,
// lane>=16: K16..31 (matches B fragment K order).
__device__ __forceinline__ v16bf load_afrag(const __bf16* rowp, int hi) {
    v16bf a = *(const v16bf*)(rowp + hi * 16);
    return a;
}
// B-matrix 32x16 bf16: lane holds column N=lane&15; lanes 0..15 carry K0..15,
// lanes 16..31 carry K16..31 (16 consecutive K values per lane).
__device__ __forceinline__ v8f wmma_bf16(v16bf a, v16bf b, v8f c) {
    return __builtin_amdgcn_wmma_f32_16x16x32_bf16(false, a, false, b, (short)0, c,
                                                   false, false);
}

// ---------- kernel 0: fp32 -> bf16 conversion ----------
__global__ void cvt_bf16_kernel(const float* __restrict__ src,
                                __bf16* __restrict__ dst, int n) {
    int i = blockIdx.x * 256 + threadIdx.x;
    if (i < n) dst[i] = f2bf(src[i]);
}

// ---------- kernel 1: RMSNorm -> bf16 ----------
__global__ __launch_bounds__(256) void rmsnorm_kernel(const float* __restrict__ x,
                                                      const float* __restrict__ w,
                                                      __bf16* __restrict__ xn) {
    int row = blockIdx.x;                    // 0..4095
    int tid = threadIdx.x;
    const float* xr = x + (long)row * DMODEL;
    float ss = 0.f;
    for (int i = tid; i < DMODEL; i += 256) { float v = xr[i]; ss += v * v; }
#pragma unroll
    for (int off = 16; off > 0; off >>= 1) ss += __shfl_xor(ss, off, 32);
    __shared__ float red[8];
    int wave = tid >> 5, lane = tid & 31;
    if (lane == 0) red[wave] = ss;
    __syncthreads();
    if (tid == 0) {
        float t = 0.f;
        for (int i = 0; i < 8; i++) t += red[i];
        red[0] = rsqrtf(t / (float)DMODEL + 1e-6f);
    }
    __syncthreads();
    float r = red[0];
    for (int i = tid; i < DMODEL; i += 256)
        xn[(long)row * DMODEL + i] = f2bf(xr[i] * r * w[i]);
}

// ---------- kernel 2: WMMA GEMM  C[M,N] = A[M,768] * W[N,768]^T ----------
// Block tile 128x128, K-step 32, double-buffered LDS fed by async LDS DMA
// (4 async b128 per thread per tile; s_wait_asynccnt 4 retires exactly the
// previous tile while the next streams underneath the WMMAs).
// 8 waves: wave (wm,wn) owns 32x64 -> 8 accumulators, 8 WMMAs per K-step.
// mode 0: store bf16 into (B,H,S,64) layout (QKV projections)
// mode 1: store fp32 = resid + acc into row-major [M,768] (output proj + residual)
__global__ __launch_bounds__(256) void gemm_kernel(const __bf16* __restrict__ A,
                                                   const __bf16* __restrict__ W,
                                                   __bf16* __restrict__ outb,
                                                   float* __restrict__ outf,
                                                   const float* __restrict__ resid,
                                                   int mode) {
    __shared__ __bf16 As[2][128][32];
    __shared__ __bf16 Bs[2][128][32];
    int tid = threadIdx.x;
    int wave = tid >> 5, lane = tid & 31;
    int hi = lane >> 4, ln = lane & 15;
    int wm = wave >> 1;        // 0..3 : 32-row strip
    int wn = wave & 1;         // 0..1 : 64-col strip (4 tiles of 16)
    int m0 = blockIdx.x * 128;
    int n0 = blockIdx.y * 128;

    // staging: each tile = 512 b128 chunks per matrix; 2 chunks/thread/matrix
    int sr = tid >> 2;            // 0..63 (also +64)
    int sc = (tid & 3) * 8;       // 0,8,16,24

    v8f acc[2][4];
#pragma unroll
    for (int r = 0; r < 2; r++)
#pragma unroll
        for (int t = 0; t < 4; t++) acc[r][t] = (v8f){};

    // prologue: stage tile 0
    async_b128(&A[(long)(m0 + sr) * DMODEL + sc],       &As[0][sr][sc]);
    async_b128(&A[(long)(m0 + sr + 64) * DMODEL + sc],  &As[0][sr + 64][sc]);
    async_b128(&W[(long)(n0 + sr) * DMODEL + sc],       &Bs[0][sr][sc]);
    async_b128(&W[(long)(n0 + sr + 64) * DMODEL + sc],  &Bs[0][sr + 64][sc]);

    int buf = 0;
    for (int k0 = 0; k0 < DMODEL; k0 += 32) {
        if (k0 + 32 < DMODEL) {
            int nb = buf ^ 1, k1 = k0 + 32;
            async_b128(&A[(long)(m0 + sr) * DMODEL + k1 + sc],      &As[nb][sr][sc]);
            async_b128(&A[(long)(m0 + sr + 64) * DMODEL + k1 + sc], &As[nb][sr + 64][sc]);
            async_b128(&W[(long)(n0 + sr) * DMODEL + k1 + sc],      &Bs[nb][sr][sc]);
            async_b128(&W[(long)(n0 + sr + 64) * DMODEL + k1 + sc], &Bs[nb][sr + 64][sc]);
            wait_async4();
        } else {
            wait_async0();
        }
        __syncthreads();

        v16bf a0 = load_afrag(&As[buf][wm * 32 + ln][0], hi);
        v16bf a1 = load_afrag(&As[buf][wm * 32 + 16 + ln][0], hi);
#pragma unroll
        for (int t = 0; t < 4; t++) {
            v16bf b = *(const v16bf*)&Bs[buf][wn * 64 + t * 16 + ln][hi * 16];
            acc[0][t] = wmma_bf16(a0, b, acc[0][t]);
            acc[1][t] = wmma_bf16(a1, b, acc[1][t]);
        }
        __syncthreads();
        buf ^= 1;
    }

    // C/D layout: lane holds column N=ln, rows M = hi*8 + v
#pragma unroll
    for (int r = 0; r < 2; r++) {
#pragma unroll
        for (int t = 0; t < 4; t++) {
#pragma unroll
            for (int v = 0; v < 8; v++) {
                int row = m0 + wm * 32 + r * 16 + hi * 8 + v;
                int col = n0 + wn * 64 + t * 16 + ln;
                if (mode == 0) {
                    int bb = row >> 11, ssq = row & (SEQ - 1);
                    int h0 = col >> 6, d0 = col & 63;
                    outb[(((long)bb * NHEADS + h0) * SEQ + ssq) * DK + d0] =
                        f2bf(acc[r][t][v]);
                } else {
                    outf[(long)row * DMODEL + col] =
                        resid[(long)row * DMODEL + col] + acc[r][t][v];
                }
            }
        }
    }
}

// ---------- kernel 3: in-place RoPE on bf16 Q and K, layout (B,H,S,64) ----------
__global__ __launch_bounds__(256) void rope_kernel(const int* __restrict__ pos,
                                                   __bf16* __restrict__ Qb,
                                                   __bf16* __restrict__ Kb) {
    long row = (long)blockIdx.x * 4 + (threadIdx.x >> 6);  // index into B*H*S
    int d = threadIdx.x & 63;
    long sq = row % SEQ;
    long bh = row / SEQ;
    long b = bh / NHEADS;
    float p = (float)pos[b * SEQ + sq];
    int dm = d & 31;
    float freq = __expf(-(float)(2 * dm) * (1.0f / 64.0f) * __logf(10000.0f));
    float ang = p * freq;
    float sn, c;
    __sincosf(ang, &sn, &c);
    long base = row * DK;
    float q  = bf2f(Qb[base + d]);
    float qo = bf2f(Qb[base + (d ^ 32)]);
    float k  = bf2f(Kb[base + d]);
    float ko = bf2f(Kb[base + (d ^ 32)]);
    float qr = (d < 32) ? (q * c - qo * sn) : (q * c + qo * sn);
    float kr = (d < 32) ? (k * c - ko * sn) : (k * c + ko * sn);
    __syncthreads();   // all readers done before in-place writes of paired lanes
    Qb[base + d] = f2bf(qr);
    Kb[base + d] = f2bf(kr);
}

// ---------- kernel 4: flash-style attention with WMMA ----------
// grid.x = SEQ/64 (query chunk), grid.y = B*H. Unscaled scores (per reference).
// K staged row-major via async LDS DMA; V staged TRANSPOSED so P*V B-fragments
// are contiguous ds_load_b128 instead of scalar column gathers.
__global__ __launch_bounds__(256) void attn_kernel(const __bf16* __restrict__ Qb,
                                                   const __bf16* __restrict__ Kb,
                                                   const __bf16* __restrict__ Vb,
                                                   const float* __restrict__ mask,
                                                   __bf16* __restrict__ attn_out) {
    __shared__ __bf16 KT[64][64];    // [key][d]
    __shared__ __bf16 VTt[64][64];   // [d][key]  (transposed V)
    __shared__ float  SC[64][64];
    __shared__ __bf16 PT[64][64];
    __shared__ float  mrow[64], lrow[64], arow[64];

    int bh = blockIdx.y;
    int b = bh / NHEADS, h = bh % NHEADS;
    int q0 = blockIdx.x * 64;
    const __bf16* Qp = Qb + ((long)bh * SEQ + q0) * DK;
    const __bf16* Kp = Kb + (long)bh * SEQ * DK;
    const __bf16* Vp = Vb + (long)bh * SEQ * DK;

    int tid = threadIdx.x;
    int wave = tid >> 5, lane = tid & 31;
    int hi = lane >> 4, ln = lane & 15;
    int wm = wave >> 1, wn = wave & 1;

    // loop-invariant Q A-fragments (rows wm*16+ln, K-dim 64 -> two frags)
    const __bf16* qrow = Qp + (long)(wm * 16 + ln) * DK;
    v16bf aq0 = load_afrag(qrow, hi);
    v16bf aq1 = load_afrag(qrow + 32, hi);

    if (tid < 64) { mrow[tid] = -1e30f; lrow[tid] = 0.f; }
    v8f pv0 = {}, pv1 = {};

    int kr0 = tid >> 3, ko0 = (tid & 7) * 8;   // K: 512 b128 chunks, 2/thread
    int lr = tid >> 2, lc = (tid & 3) * 16;    // V: 16 bf16 per thread

    for (int kt = 0; kt < SEQ; kt += 64) {
        // K tile via async LDS DMA
        async_b128(&Kp[(long)(kt + kr0) * DK + ko0], &KT[kr0][ko0]);
        async_b128(&Kp[(long)(kt + kr0 + 32) * DK + ko0], &KT[kr0 + 32][ko0]);
        // V tile loaded and transposed into LDS
        v16bf vv = *(const v16bf*)&Vp[(long)(kt + lr) * DK + lc];
#pragma unroll
        for (int i = 0; i < 16; i++) VTt[lc + i][lr] = vv[i];
        wait_async0();
        __syncthreads();

        // scores S = Q * K^T  (B[k][n] = K[n][k] -> contiguous in KT rows)
        v8f s0 = {}, s1 = {};
        {
            v16bf b0 = *(const v16bf*)&KT[wn * 32 + ln][hi * 16];
            v16bf b1 = *(const v16bf*)&KT[wn * 32 + 16 + ln][hi * 16];
            s0 = wmma_bf16(aq0, b0, s0);
            s1 = wmma_bf16(aq0, b1, s1);
            b0 = *(const v16bf*)&KT[wn * 32 + ln][32 + hi * 16];
            b1 = *(const v16bf*)&KT[wn * 32 + 16 + ln][32 + hi * 16];
            s0 = wmma_bf16(aq1, b0, s0);
            s1 = wmma_bf16(aq1, b1, s1);
        }
#pragma unroll
        for (int v = 0; v < 8; v++) {
            SC[wm * 16 + hi * 8 + v][wn * 32 + ln] = s0[v];
            SC[wm * 16 + hi * 8 + v][wn * 32 + 16 + ln] = s1[v];
        }
        __syncthreads();

        // fp32 online softmax per query row (threads 0..63)
        if (tid < 64) {
            const float* mr = mask + ((long)b * SEQ + q0 + tid) * SEQ + kt;
            float mold = mrow[tid];
            float tmax = -1e30f;
            for (int j = 0; j < 64; j++) {
                float v = SC[tid][j] + mr[j];
                SC[tid][j] = v;
                tmax = fmaxf(tmax, v);
            }
            float mnew = fmaxf(mold, tmax);
            float alpha = __expf(mold - mnew);
            float sum = 0.f;
            for (int j = 0; j < 64; j++) {
                float pv = __expf(SC[tid][j] - mnew);
                PT[tid][j] = f2bf(pv);
                sum += pv;
            }
            lrow[tid] = alpha * lrow[tid] + sum;
            arow[tid] = alpha;
            mrow[tid] = mnew;
        }
        __syncthreads();

        // rescale accumulators, then PV += P * V
#pragma unroll
        for (int v = 0; v < 8; v++) {
            float sc = arow[wm * 16 + hi * 8 + v];
            pv0[v] *= sc;
            pv1[v] *= sc;
        }
        const __bf16* prow = &PT[wm * 16 + ln][0];
        v16bf ap0 = load_afrag(prow, hi);
        v16bf ap1 = load_afrag(prow + 32, hi);
        // B frag elem i = V[kk+i][n] = VTt[n][kk+i] -> contiguous b128 pairs
        v16bf bv0 = *(const v16bf*)&VTt[wn * 32 + ln][hi * 16];
        v16bf bv1 = *(const v16bf*)&VTt[wn * 32 + 16 + ln][hi * 16];
        pv0 = wmma_bf16(ap0, bv0, pv0);
        pv1 = wmma_bf16(ap0, bv1, pv1);
        bv0 = *(const v16bf*)&VTt[wn * 32 + ln][32 + hi * 16];
        bv1 = *(const v16bf*)&VTt[wn * 32 + 16 + ln][32 + hi * 16];
        pv0 = wmma_bf16(ap1, bv0, pv0);
        pv1 = wmma_bf16(ap1, bv1, pv1);
        __syncthreads();
    }

    // normalize and emit bf16 attn in (B,S,INNER) layout for the output GEMM
#pragma unroll
    for (int v = 0; v < 8; v++) {
        int m = wm * 16 + hi * 8 + v;
        float inv = 1.0f / lrow[m];
        long base = ((long)b * SEQ + q0 + m) * DMODEL + h * DK;
        attn_out[base + wn * 32 + ln]      = f2bf(pv0[v] * inv);
        attn_out[base + wn * 32 + 16 + ln] = f2bf(pv1[v] * inv);
    }
}

// ---------- host-side orchestration ----------
extern "C" void kernel_launch(void* const* d_in, const int* in_sizes, int n_in,
                              void* d_out, int out_size, void* d_ws, size_t ws_size,
                              hipStream_t stream) {
    const float* hidden = (const float*)d_in[0];
    const float* mask   = (const float*)d_in[1];
    const int*   pos    = (const int*)d_in[2];
    const float* lnw    = (const float*)d_in[3];
    const float* wq     = (const float*)d_in[4];
    const float* wk     = (const float*)d_in[5];
    const float* wv     = (const float*)d_in[6];
    const float* wo     = (const float*)d_in[7];
    float* out = (float*)d_out;

    char* p = (char*)d_ws;
    auto alloc = [&](size_t bytes) -> void* {
        void* r = (void*)p;
        p += (bytes + 255) & ~(size_t)255;
        return r;
    };
    const size_t WBYTES = (size_t)DMODEL * DMODEL * sizeof(__bf16);
    const size_t XBYTES = (size_t)MTOT * DMODEL * sizeof(__bf16);
    __bf16* Wqb = (__bf16*)alloc(WBYTES);
    __bf16* Wkb = (__bf16*)alloc(WBYTES);
    __bf16* Wvb = (__bf16*)alloc(WBYTES);
    __bf16* Wob = (__bf16*)alloc(WBYTES);
    __bf16* Xn  = (__bf16*)alloc(XBYTES);
    __bf16* Qb  = (__bf16*)alloc(XBYTES);   // (B,H,S,64)
    __bf16* Kb  = (__bf16*)alloc(XBYTES);
    __bf16* Vb  = (__bf16*)alloc(XBYTES);
    __bf16* Ab  = (__bf16*)alloc(XBYTES);   // attn result (B,S,768)

    const int WN = DMODEL * DMODEL;
    cvt_bf16_kernel<<<(WN + 255) / 256, 256, 0, stream>>>(wq, Wqb, WN);
    cvt_bf16_kernel<<<(WN + 255) / 256, 256, 0, stream>>>(wk, Wkb, WN);
    cvt_bf16_kernel<<<(WN + 255) / 256, 256, 0, stream>>>(wv, Wvb, WN);
    cvt_bf16_kernel<<<(WN + 255) / 256, 256, 0, stream>>>(wo, Wob, WN);

    rmsnorm_kernel<<<MTOT, 256, 0, stream>>>(hidden, lnw, Xn);

    dim3 ggrid(MTOT / 128, DMODEL / 128);   // 32 x 6
    gemm_kernel<<<ggrid, 256, 0, stream>>>(Xn, Wqb, Qb, nullptr, nullptr, 0);
    gemm_kernel<<<ggrid, 256, 0, stream>>>(Xn, Wkb, Kb, nullptr, nullptr, 0);
    gemm_kernel<<<ggrid, 256, 0, stream>>>(Xn, Wvb, Vb, nullptr, nullptr, 0);

    rope_kernel<<<(BATCH * NHEADS * SEQ) / 4, 256, 0, stream>>>(pos, Qb, Kb);

    dim3 agrid(SEQ / 64, BATCH * NHEADS);
    attn_kernel<<<agrid, 256, 0, stream>>>(Qb, Kb, Vb, mask, Ab);

    gemm_kernel<<<ggrid, 256, 0, stream>>>(Ab, Wob, nullptr, out, hidden, 1);
}
